// MoEGate_2877628088861
// MI455X (gfx1250) — matrix operand order repaired and checked by reading, not compiled, not run on hardware
//
#include <hip/hip_runtime.h>
#include <hip/hip_bf16.h>

typedef __attribute__((ext_vector_type(16))) __bf16 v16bf;
typedef __attribute__((ext_vector_type(8)))  float  v8f;
typedef __attribute__((ext_vector_type(4)))  float  f4;

constexpr int H = 2048;   // hidden dim (K)
constexpr int E = 64;     // experts (N)
constexpr int TOPK = 8;
constexpr int RS = 68;    // LDS logits row stride (floats): 16B-aligned rows, conflict-unfriendly phase
constexpr int WPB = 8;    // waves per block
constexpr int TILES_PER_BLOCK = 4;  // 2 k-slice waves per tile

// ---- tiny pre-pass: fp32 weight -> bf16 in workspace (512KB -> 256KB) ----
__global__ __launch_bounds__(256) void convert_weight_kernel(
    const float* __restrict__ w, __bf16* __restrict__ wb, int n) {
  int i = blockIdx.x * 256 + threadIdx.x;
  if (i < n) wb[i] = (__bf16)w[i];
}

__device__ __forceinline__ void load_a_raw(const float* pa, f4& x0, f4& x1, f4& x2, f4& x3) {
  // A 16x32 bf16 layout: elems 0..15 at k+half*16 (contiguous)
  x0 = __builtin_nontemporal_load((const f4*)(pa));       // streamed once: NT hint
  x1 = __builtin_nontemporal_load((const f4*)(pa + 4));
  x2 = __builtin_nontemporal_load((const f4*)(pa + 8));
  x3 = __builtin_nontemporal_load((const f4*)(pa + 12));
}

__device__ __forceinline__ v16bf cvt_a(f4 x0, f4 x1, f4 x2, f4 x3) {
  v16bf a;
#pragma unroll
  for (int i = 0; i < 4; ++i) {
    a[i]      = (__bf16)x0[i];
    a[4 + i]  = (__bf16)x1[i];
    a[8 + i]  = (__bf16)x2[i];
    a[12 + i] = (__bf16)x3[i];
  }
  return a;
}

__global__ __launch_bounds__(256) void moe_gate_kernel(
    const float* __restrict__ hidden,   // [T, H] fp32
    const __bf16* __restrict__ wbf,     // [E, H] bf16
    float* __restrict__ out,            // [T*8] weights ++ [T*8] indices(as float)
    int T) {
  __shared__ float slog[WPB * 16 * RS];               // per-wave partial logits
  __shared__ float scv[TILES_PER_BLOCK * 16 * 16];    // per-tile candidate values
  __shared__ int   sci[TILES_PER_BLOCK * 16 * 16];    // per-tile candidate indices

  const int lane   = threadIdx.x & 31;
  const int wave   = threadIdx.x >> 5;
  const int col    = lane & 15;   // token row (A) / expert col (B) in tile
  const int half   = lane >> 4;
  const int tileIB = wave >> 1;   // tile within block
  const int kslice = wave & 1;    // K-split: 2 waves per tile

  const int tile   = blockIdx.x * TILES_PER_BLOCK + tileIB;
  const int token0 = tile * 16;

  const int k0   = kslice * (H / 2);
  const int kend = k0 + (H / 2);

  const float*  arow = hidden + (size_t)(token0 + col) * H;
  const __bf16* brow = wbf + (size_t)col * H + half * 16;  // lane=expert col, 16 contiguous K per half

  v8f acc0 = {}, acc1 = {}, acc2 = {}, acc3 = {};

  // ---- software-pipelined K loop: prefetch next A(raw fp32) + 4 B frags ----
  f4 xa0, xa1, xa2, xa3, na0, na1, na2, na3;
  v16bf b0, b1, b2, b3, nb0, nb1, nb2, nb3;
  load_a_raw(arow + k0 + half * 16, xa0, xa1, xa2, xa3);
  b0 = *(const v16bf*)(brow + k0);
  b1 = *(const v16bf*)(brow + (size_t)16 * H + k0);
  b2 = *(const v16bf*)(brow + (size_t)32 * H + k0);
  b3 = *(const v16bf*)(brow + (size_t)48 * H + k0);

  for (int kk = k0; kk < kend; kk += 32) {
    const int nk = (kk + 32 < kend) ? kk + 32 : k0;   // clamp avoids OOB on last step
    load_a_raw(arow + nk + half * 16, na0, na1, na2, na3);
    nb0 = *(const v16bf*)(brow + nk);
    nb1 = *(const v16bf*)(brow + (size_t)16 * H + nk);
    nb2 = *(const v16bf*)(brow + (size_t)32 * H + nk);
    nb3 = *(const v16bf*)(brow + (size_t)48 * H + nk);

    v16bf a = cvt_a(xa0, xa1, xa2, xa3);
    acc0 = __builtin_amdgcn_wmma_f32_16x16x32_bf16(false, a, false, b0, (short)0, acc0, false, false);
    acc1 = __builtin_amdgcn_wmma_f32_16x16x32_bf16(false, a, false, b1, (short)0, acc1, false, false);
    acc2 = __builtin_amdgcn_wmma_f32_16x16x32_bf16(false, a, false, b2, (short)0, acc2, false, false);
    acc3 = __builtin_amdgcn_wmma_f32_16x16x32_bf16(false, a, false, b3, (short)0, acc3, false, false);

    xa0 = na0; xa1 = na1; xa2 = na2; xa3 = na3;
    b0 = nb0; b1 = nb1; b2 = nb2; b3 = nb3;
  }

  // ---- scatter partial logits: C layout lane l, vgpr r -> (m=r+8*half, n=col) ----
  float* myl = slog + wave * 16 * RS;
#pragma unroll
  for (int r = 0; r < 8; ++r) {
    float* mrow = myl + (r + 8 * half) * RS;
    mrow[col]      = acc0[r];
    mrow[col + 16] = acc1[r];
    mrow[col + 32] = acc2[r];
    mrow[col + 48] = acc3[r];
  }
  __syncthreads();   // k-slice partner wave must finish its scatter

  if (kslice == 0) {
    // ---- phase 2: all 32 lanes; lane = (token=col, expert-half=half) ----
    // combine the two K-partials during load, then register-resident top-8-of-32
    const float* p0 = slog + (wave + 0) * 16 * RS + col * RS + half * 32;
    const float* p1 = slog + (wave + 1) * 16 * RS + col * RS + half * 32;
    float vals[32];
#pragma unroll
    for (int j = 0; j < 8; ++j) {
      f4 u = *(const f4*)(p0 + 4 * j);
      f4 v = *(const f4*)(p1 + 4 * j);
#pragma unroll
      for (int i = 0; i < 4; ++i) vals[4 * j + i] = u[i] + v[i];
    }
    unsigned used = 0u;
    float* cw = scv + tileIB * 256 + col * 16 + half * 8;
    int*   ci = sci + tileIB * 256 + col * 16 + half * 8;
#pragma unroll 1
    for (int s = 0; s < TOPK; ++s) {
      float mx = -3.402823466e+38f;
      int   mi = 0;
#pragma unroll
      for (int j = 0; j < 32; ++j) {
        bool take = (((used >> j) & 1u) == 0u) && (vals[j] > mx);
        mx = take ? vals[j] : mx;
        mi = take ? j : mi;
      }
      used |= (1u << mi);
      cw[s] = mx;
      ci[s] = mi + half * 32;   // global expert index
    }

    // ---- phase 3: lanes 0..15 merge the two sorted 8-candidate lists ----
    if (lane < 16) {
      float cv[16];
      int   cx[16];
      const float* rv = scv + tileIB * 256 + lane * 16;
      const int*   ri = sci + tileIB * 256 + lane * 16;
#pragma unroll
      for (int j = 0; j < 16; ++j) { cv[j] = rv[j]; cx[j] = ri[j]; }

      unsigned used16 = 0u;
      float bw[TOPK];
      int   bi[TOPK];
#pragma unroll 1
      for (int s = 0; s < TOPK; ++s) {
        float mx = -3.402823466e+38f;
        int   mi = 0;
#pragma unroll
        for (int j = 0; j < 16; ++j) {   // scan order = half0 desc, half1 desc -> lowest-index tie-break
          bool take = (((used16 >> j) & 1u) == 0u) && (cv[j] > mx);
          mx = take ? cv[j] : mx;
          mi = take ? j : mi;
        }
        used16 |= (1u << mi);
        bw[s] = mx;
        bi[s] = cx[mi];
      }

      // global softmax Z cancels under top-k renormalization
      float m0 = bw[0];
      float e[TOPK];
      float sum = 0.f;
#pragma unroll
      for (int s = 0; s < TOPK; ++s) { e[s] = __expf(bw[s] - m0); sum += e[s]; }
      float inv = 1.0f / sum;

      const int token = token0 + lane;
      float* ow = out + (size_t)token * TOPK;
      float* oi = out + (size_t)T * TOPK + (size_t)token * TOPK;
#pragma unroll
      for (int s = 0; s < TOPK; ++s) {
        ow[s] = e[s] * inv;
        oi[s] = (float)bi[s];
      }
    }
  }
}

extern "C" void kernel_launch(void* const* d_in, const int* in_sizes, int n_in,
                              void* d_out, int out_size, void* d_ws, size_t ws_size,
                              hipStream_t stream) {
  const float* hidden = (const float*)d_in[0];   // [4,4096,2048] fp32
  const float* weight = (const float*)d_in[1];   // [64,2048] fp32
  float* out = (float*)d_out;
  __bf16* wbf = (__bf16*)d_ws;                   // 256 KB bf16 weight staging

  const int T  = in_sizes[0] / H;                // 16384 tokens
  const int NW = E * H;                          // 131072 weight elems

  convert_weight_kernel<<<(NW + 255) / 256, 256, 0, stream>>>(weight, wbf, NW);

  const int tiles  = T / 16;                     // 1024 tiles
  const int blocks = tiles / TILES_PER_BLOCK;    // 256 blocks x 8 waves = 2048 waves
  moe_gate_kernel<<<blocks, 256, 0, stream>>>(hidden, wbf, out, T);
}
